// HiResPrecipNet_9x_25x_CNN_1563368096253
// MI455X (gfx1250) — matrix operand order repaired and checked by reading, not compiled, not run on hardware
//
#include <hip/hip_runtime.h>
#include <hip/hip_bf16.h>
#include <math.h>

// ---------------------------------------------------------------------------
// HiResPrecipNet (CNN + GATv2 stack) for gfx1250 / MI455X.
// Dense projections: bf16 WMMA (v_wmma_f32_16x16x32_bf16, f32 accumulate),
// fed by pre-padded bf16 activations [M,Kp] and pre-transposed bf16 weights
// [N,Kp]; inner loop is software-pipelined (last tile peeled) so the steady
// state is unconditional b128 loads overlapped with the matrix pipe.
// Edge softmax / scatter-mean: atomic segment ops (HBM bound).
// ---------------------------------------------------------------------------

typedef __bf16 bf16_t;
typedef __attribute__((ext_vector_type(8)))  __bf16 v8bf;
typedef __attribute__((ext_vector_type(16))) __bf16 v16bf;
typedef __attribute__((ext_vector_type(8)))  float  v8f;

#define WG 256
#define CC 64                 // GAT head width (outc) is 64 everywhere

#define N_LOW 6000
#define N9    40000
#define N25   100000
#define NH    60000
#define E1    360000
#define E2    900000
#define E3    540000
#define EHH   600000

__device__ __forceinline__ void atomicMaxF(float* addr, float val) {
  if (val >= 0.0f) atomicMax((int*)addr, __float_as_int(val));
  else             atomicMin((unsigned int*)addr, (unsigned int)__float_as_int(val));
}

// --------------------------- bf16 staging ----------------------------------
// f32 [M,K] -> bf16 [M,Kp], zero-padded K tail (Kp % 32 == 0)
__global__ void k_cvt_pad(const float* __restrict__ x, bf16_t* __restrict__ y,
                          int M, int K, int Kp) {
  int i = blockIdx.x * blockDim.x + threadIdx.x;
  if (i >= M * Kp) return;
  int m = i / Kp, k = i - m * Kp;
  y[i] = (k < K) ? (bf16_t)x[(size_t)m * K + k] : (bf16_t)0.0f;
}

// f32 B [K,N] -> bf16 Bt [N,Kp] (transposed, zero-padded)
__global__ void k_cvt_transpose(const float* __restrict__ B, bf16_t* __restrict__ Bt,
                                int K, int N, int Kp) {
  int i = blockIdx.x * blockDim.x + threadIdx.x;
  if (i >= N * Kp) return;
  int n = i / Kp, k = i - n * Kp;
  Bt[i] = (k < K) ? (bf16_t)B[(size_t)k * N + n] : (bf16_t)0.0f;
}

// --------------------------- WMMA GEMM -------------------------------------
// C[M,N] = A[M,Kp](bf16) @ Bt[N,Kp](bf16)^T + bias, optional ReLU.
// One wave -> 16x32 C slab (2 wmma accumulators); 4 waves/block tile M by 64.
// M % 16 == 0 and N % 32 == 0 by construction: no bounds checks in hot loop.
// Pipelined with the last K-tile peeled: loop body loads tile k+1
// unconditionally, then issues wmma on tile k (partial loadcnt waits).
__global__ void k_gemm_bf16_wmma(const bf16_t* __restrict__ A,
                                 const bf16_t* __restrict__ Bt,
                                 const float* __restrict__ bias,
                                 float* __restrict__ C,
                                 int M, int N, int Kp, int relu) {
  const int lane = threadIdx.x & 31;
  const int wave = threadIdx.x >> 5;
  const int m0 = (blockIdx.y * 4 + wave) * 16;
  if (m0 >= M) return;                    // wave-uniform guard, EXEC stays full
  const int n0 = blockIdx.x * 32;
  const int r    = lane & 15;
  const int half = lane >> 4;

  const bf16_t* pa  = A  + (size_t)(m0 + r) * Kp + half * 16;  // A frag runs
  const bf16_t* pb0 = Bt + (size_t)(n0 + r) * Kp + half * 16;  // B frag col n0+r
  const bf16_t* pb1 = Bt + (size_t)(n0 + 16 + r) * Kp + half * 16;

  // prologue: first K-tile in flight
  v8bf  alo = *(const v8bf*)(pa);
  v8bf  ahi = *(const v8bf*)(pa + 8);
  v16bf b0  = *(const v16bf*)(pb0);
  v16bf b1  = *(const v16bf*)(pb1);

  v8f acc0 = {}, acc1 = {};
  int k0 = 0;
  for (; k0 + 32 < Kp; k0 += 32) {
    // A fragment 16x32: elems 0..15 -> K=k0+16*half+{0..15}
    v16bf a = __builtin_shufflevector(alo, ahi, 0, 1, 2, 3, 4, 5, 6, 7,
                                      8, 9, 10, 11, 12, 13, 14, 15);
    v16bf c0 = b0, c1 = b1;
    // unconditionally fetch next tile before consuming current one
    alo = *(const v8bf*)(pa + k0 + 32);
    ahi = *(const v8bf*)(pa + k0 + 40);
    b0  = *(const v16bf*)(pb0 + k0 + 32);
    b1  = *(const v16bf*)(pb1 + k0 + 32);
    acc0 = __builtin_amdgcn_wmma_f32_16x16x32_bf16(false, a, false, c0,
                                                   (short)0, acc0, false, false);
    acc1 = __builtin_amdgcn_wmma_f32_16x16x32_bf16(false, a, false, c1,
                                                   (short)0, acc1, false, false);
  }
  // peeled last tile
  {
    v16bf a = __builtin_shufflevector(alo, ahi, 0, 1, 2, 3, 4, 5, 6, 7,
                                      8, 9, 10, 11, 12, 13, 14, 15);
    acc0 = __builtin_amdgcn_wmma_f32_16x16x32_bf16(false, a, false, b0,
                                                   (short)0, acc0, false, false);
    acc1 = __builtin_amdgcn_wmma_f32_16x16x32_bf16(false, a, false, b1,
                                                   (short)0, acc1, false, false);
  }
  const int cn0 = n0 + r, cn1 = n0 + 16 + r;
  const float bz0 = bias ? bias[cn0] : 0.0f;
  const float bz1 = bias ? bias[cn1] : 0.0f;
#pragma unroll
  for (int i = 0; i < 8; ++i) {
    int cm = m0 + i + 8 * half;           // D layout: VGPR i -> row i (+8 upper)
    float v0 = acc0[i] + bz0;
    float v1 = acc1[i] + bz1;
    if (relu) { v0 = fmaxf(v0, 0.0f); v1 = fmaxf(v1, 0.0f); }
    C[(size_t)cm * N + cn0] = v0;
    C[(size_t)cm * N + cn1] = v1;
  }
}

// ----------------------------- utility kernels -----------------------------
__global__ void k_fill(float* __restrict__ p, float v, int n) {
  int i = blockIdx.x * blockDim.x + threadIdx.x;
  if (i < n) p[i] = v;
}

__global__ void k_make_edges(const int* __restrict__ ehh, int* __restrict__ s2,
                             int* __restrict__ d2, int E, int N) {
  int i = blockIdx.x * blockDim.x + threadIdx.x;
  if (i >= E + N) return;
  if (i < E) { s2[i] = ehh[i]; d2[i] = ehh[E + i]; }
  else       { s2[i] = i - E;  d2[i] = i - E; }
}

__global__ void k_concat(const float* __restrict__ z, const float* __restrict__ h,
                         float* __restrict__ xc, int N) {
  int i = blockIdx.x * blockDim.x + threadIdx.x;
  if (i >= N * 65) return;
  int n = i / 65, c = i - n * 65;
  xc[i] = (c == 0) ? z[n] : h[(size_t)n * 64 + c - 1];
}

// ------------------------------- GATv2 edge ops ----------------------------
__global__ void k_edge_logits(const float* __restrict__ xl, const float* __restrict__ xr,
                              const int* __restrict__ src, const int* __restrict__ dst,
                              const float* __restrict__ att, float* __restrict__ elog,
                              float* __restrict__ m, int E, int H) {
  int idx = blockIdx.x * blockDim.x + threadIdx.x;
  if (idx >= E * H) return;
  int e = idx / H, h = idx - e * H;
  int s = src[e], d = dst[e];
  const float4* pl = (const float4*)(xl + (size_t)s * H * CC + h * CC);
  const float4* pr = (const float4*)(xr + (size_t)d * H * CC + h * CC);
  const float4* pa = (const float4*)(att + h * CC);
  float acc = 0.0f;
#pragma unroll 4
  for (int c = 0; c < CC / 4; ++c) {
    float4 a = pl[c], b = pr[c], w = pa[c];
    float v;
    v = a.x + b.x; v = (v > 0.0f) ? v : 0.2f * v; acc += w.x * v;
    v = a.y + b.y; v = (v > 0.0f) ? v : 0.2f * v; acc += w.y * v;
    v = a.z + b.z; v = (v > 0.0f) ? v : 0.2f * v; acc += w.z * v;
    v = a.w + b.w; v = (v > 0.0f) ? v : 0.2f * v; acc += w.w * v;
  }
  elog[idx] = acc;
  atomicMaxF(&m[(size_t)d * H + h], acc);
}

__global__ void k_edge_exp(float* __restrict__ elog, const float* __restrict__ m,
                           float* __restrict__ ssum, float* __restrict__ cnt,
                           const int* __restrict__ dst, int E, int H) {
  int idx = blockIdx.x * blockDim.x + threadIdx.x;
  if (idx >= E * H) return;
  int e = idx / H, h = idx - e * H;
  int d = dst[e];
  float a = __expf(elog[idx] - m[(size_t)d * H + h]);
  elog[idx] = a;
  atomicAdd(&ssum[(size_t)d * H + h], a);
  if (h == 0) atomicAdd(&cnt[d], 1.0f);
}

__global__ void k_edge_scatter(const float* __restrict__ xl, const float* __restrict__ ea,
                               const float* __restrict__ ssum,
                               const int* __restrict__ src, const int* __restrict__ dst,
                               float* __restrict__ agg, int E, int H) {
  int idx = blockIdx.x * blockDim.x + threadIdx.x;
  if (idx >= E * H) return;
  int e = idx / H, h = idx - e * H;
  int s = src[e], d = dst[e];
  float alpha = ea[idx] / ssum[(size_t)d * H + h];
  const float* pl = xl + (size_t)s * H * CC + h * CC;
  float* pd = agg + (size_t)d * H * CC + h * CC;
#pragma unroll 8
  for (int c = 0; c < CC; ++c) atomicAdd(&pd[c], pl[c] * alpha);
}

__global__ void k_gat_finish(const float* __restrict__ agg, const float* __restrict__ cnt,
                             const float* __restrict__ b, float* __restrict__ out,
                             int N, int HC, int relu) {
  int i = blockIdx.x * blockDim.x + threadIdx.x;
  if (i >= N * HC) return;
  int n = i / HC, f = i - n * HC;
  float v = agg[i] / fmaxf(cnt[n], 1.0f) + b[f];
  out[i] = relu ? fmaxf(v, 0.0f) : v;
}

// ------------------------------- BatchNorm1d -------------------------------
__global__ void k_colstats(const float* __restrict__ x, float* __restrict__ stats,
                           int Nrows, int F) {
  int f = threadIdx.x;
  if (f >= F) return;
  int rows_per_block = (Nrows + gridDim.x - 1) / gridDim.x;
  int r0 = blockIdx.x * rows_per_block;
  int r1 = min(r0 + rows_per_block, Nrows);
  float s = 0.0f, s2 = 0.0f;
  for (int r = r0; r < r1; ++r) {
    float v = x[(size_t)r * F + f];
    s += v; s2 += v * v;
  }
  atomicAdd(&stats[f], s);
  atomicAdd(&stats[F + f], s2);
}

__global__ void k_bn_apply_relu(float* __restrict__ x, const float* __restrict__ stats,
                                const float* __restrict__ g, const float* __restrict__ b,
                                int N, int F) {
  int i = blockIdx.x * blockDim.x + threadIdx.x;
  if (i >= N * F) return;
  int f = i % F;
  float inv = 1.0f / (float)N;
  float mu  = stats[f] * inv;
  float var = stats[F + f] * inv - mu * mu;
  float v = (x[i] - mu) * rsqrtf(var + 1e-5f) * g[f] + b[f];
  x[i] = fmaxf(v, 0.0f);
}

// --------------------------------- CNN -------------------------------------
__global__ void k_dwconv(const float* __restrict__ x, const float* __restrict__ w,
                         const float* __restrict__ cb, float* __restrict__ y,
                         float* __restrict__ stats, int N) {
  __shared__ float lsum[5], lsq[5];
  if (threadIdx.x < 5) { lsum[threadIdx.x] = 0.0f; lsq[threadIdx.x] = 0.0f; }
  __syncthreads();
  int idx = blockIdx.x * blockDim.x + threadIdx.x;
  int total = N * 125;
  if (idx < total) {
    int n = idx / 125, rem = idx - n * 125;
    int ch = rem / 25, p = rem - ch * 25;
    int py = p / 5, px = p - py * 5;
    const float* xp = x + (size_t)n * 125 + ch * 25;
    const float* wp = w + ch * 9;
    float acc = cb[ch];
    for (int dy = -1; dy <= 1; ++dy)
      for (int dx = -1; dx <= 1; ++dx) {
        int yy = py + dy, xx = px + dx;
        if (yy >= 0 && yy < 5 && xx >= 0 && xx < 5)
          acc += wp[(dy + 1) * 3 + (dx + 1)] * xp[yy * 5 + xx];
      }
    y[idx] = acc;
    atomicAdd(&lsum[ch], acc);            // ds_add_f32 block reduction
    atomicAdd(&lsq[ch], acc * acc);
  }
  __syncthreads();
  if (threadIdx.x < 5) {
    atomicAdd(&stats[threadIdx.x], lsum[threadIdx.x]);
    atomicAdd(&stats[5 + threadIdx.x], lsq[threadIdx.x]);
  }
}

__global__ void k_cnn_bn(float* __restrict__ x, const float* __restrict__ stats,
                         const float* __restrict__ g, const float* __restrict__ bt, int N) {
  int i = blockIdx.x * blockDim.x + threadIdx.x;
  if (i >= N * 125) return;
  int ch = (i % 125) / 25;
  float inv = 1.0f / (float)(N * 25);
  float mu  = stats[ch] * inv;
  float var = stats[5 + ch] * inv - mu * mu;
  float v = (x[i] - mu) * rsqrtf(var + 1e-5f);
  x[i] = fmaxf(v * g[ch] + bt[ch], 0.0f);
}

__global__ void k_pool45(const float* __restrict__ x, float* __restrict__ out, int N) {
  int i = blockIdx.x * blockDim.x + threadIdx.x;
  if (i >= N * 45) return;
  int n = i / 45, f = i - n * 45;
  int ch = f / 9, p = f - ch * 9;
  int oy = p / 3, ox = p - oy * 3;
  const float* xp = x + (size_t)n * 125 + ch * 25;
  float m = -INFINITY;
  for (int dy = 0; dy < 2; ++dy) {
    int yy = 2 * oy - 1 + dy; if (yy < 0 || yy >= 5) continue;
    for (int dx = 0; dx < 2; ++dx) {
      int xx = 2 * ox - 1 + dx; if (xx < 0 || xx >= 5) continue;
      m = fmaxf(m, xp[yy * 5 + xx]);
    }
  }
  out[i] = m;
}

// ------------------------------- predictor tail ----------------------------
__global__ void k_dotN(const float* __restrict__ x, const float* __restrict__ w,
                       const float* __restrict__ b, float* __restrict__ out, int N, int K) {
  int n = blockIdx.x * blockDim.x + threadIdx.x;
  if (n >= N) return;
  float acc = b[0];
  for (int k = 0; k < K; ++k) acc += x[(size_t)n * K + k] * w[k];
  out[n] = acc;
}

// ------------------------------- host side ---------------------------------
struct GatBufs {
  float *xl, *xr, *elog, *m, *ssum, *cnt, *agg;
  bf16_t *abf, *abf2, *wbf, *wbf2;
};

// full f32 GEMM: stage A (pad) + B (transpose) to bf16, then WMMA
static void gemm_f32(hipStream_t s, const float* A, int M, int K,
                     const float* B, int N, const float* bias, int relu,
                     float* C, bf16_t* abf, bf16_t* wbf) {
  int Kp = (K + 31) & ~31;
  k_cvt_pad<<<((size_t)M * Kp + WG - 1) / WG, WG, 0, s>>>(A, abf, M, K, Kp);
  k_cvt_transpose<<<((size_t)N * Kp + WG - 1) / WG, WG, 0, s>>>(B, wbf, K, N, Kp);
  dim3 g(N / 32, (M + 63) / 64);
  k_gemm_bf16_wmma<<<g, 128, 0, s>>>(abf, wbf, bias, C, M, N, Kp, relu);
}

static void run_gat(hipStream_t s,
                    const float* x_src, int Ns, int Fs,
                    const float* x_dst, int Nd, int Fd,
                    const int* src, const int* dst, int E,
                    const float* Wl, const float* bl,
                    const float* Wr, const float* br,
                    const float* att, const float* bout,
                    int H, int relu_out, float* out, const GatBufs& t) {
  int HC = H * CC;
  int KpS = (Fs + 31) & ~31;
  int KpD = (Fd + 31) & ~31;
  // xl = x_src @ Wl + bl
  k_cvt_pad<<<((size_t)Ns * KpS + WG - 1) / WG, WG, 0, s>>>(x_src, t.abf, Ns, Fs, KpS);
  k_cvt_transpose<<<((size_t)HC * KpS + WG - 1) / WG, WG, 0, s>>>(Wl, t.wbf, Fs, HC, KpS);
  {
    dim3 g(HC / 32, (Ns + 63) / 64);
    k_gemm_bf16_wmma<<<g, 128, 0, s>>>(t.abf, t.wbf, bl, t.xl, Ns, HC, KpS, 0);
  }
  // xr = x_dst @ Wr + br (reuse staged A when src==dst)
  const bf16_t* adst = t.abf;
  if (!(x_src == x_dst && Fs == Fd)) {
    k_cvt_pad<<<((size_t)Nd * KpD + WG - 1) / WG, WG, 0, s>>>(x_dst, t.abf2, Nd, Fd, KpD);
    adst = t.abf2;
  }
  k_cvt_transpose<<<((size_t)HC * KpD + WG - 1) / WG, WG, 0, s>>>(Wr, t.wbf2, Fd, HC, KpD);
  {
    dim3 g(HC / 32, (Nd + 63) / 64);
    k_gemm_bf16_wmma<<<g, 128, 0, s>>>(adst, t.wbf2, br, t.xr, Nd, HC, KpD, 0);
  }
  // edge softmax + mean aggregation
  k_fill<<<(Nd * H + WG - 1) / WG, WG, 0, s>>>(t.m, -INFINITY, Nd * H);
  k_fill<<<(Nd * H + WG - 1) / WG, WG, 0, s>>>(t.ssum, 0.0f, Nd * H);
  k_fill<<<(Nd + WG - 1) / WG, WG, 0, s>>>(t.cnt, 0.0f, Nd);
  k_fill<<<(Nd * HC + WG - 1) / WG, WG, 0, s>>>(t.agg, 0.0f, Nd * HC);
  int EH = E * H;
  k_edge_logits<<<(EH + WG - 1) / WG, WG, 0, s>>>(t.xl, t.xr, src, dst, att, t.elog, t.m, E, H);
  k_edge_exp<<<(EH + WG - 1) / WG, WG, 0, s>>>(t.elog, t.m, t.ssum, t.cnt, dst, E, H);
  k_edge_scatter<<<(EH + WG - 1) / WG, WG, 0, s>>>(t.xl, t.elog, t.ssum, src, dst, t.agg, E, H);
  k_gat_finish<<<(Nd * HC + WG - 1) / WG, WG, 0, s>>>(t.agg, t.cnt, bout, out, Nd, HC, relu_out);
}

// params pytree leaves in JAX sorted-key order
static const int kParamSizes[66] = {
  15, 15, 15, 135,                        // cb, cbt, cg, cw
  2880, 64, 64, 64, 64, 64,               // d1: Wl,Wr,att,b,bl,br
  4096, 64, 64, 64, 64, 64,               // d2
  4096, 64, 64, 64, 64, 64,               // d3
  128, 128, 128, 128, 128, 128, 128, 128, // pbn: b0,g0,b1,g1,b2,g2,b3,g3
  4096, 2048, 32, 64, 32, 1,              // pred: W1,W2,W3,b1,b2,b3
  8320, 8320, 128, 128, 128, 128,         // proc0
  16384, 16384, 128, 128, 128, 128,       // proc1
  16384, 16384, 128, 128, 128, 128,       // proc2
  16384, 16384, 128, 128, 128, 128,       // proc3
  8192, 8192, 64, 64, 64, 64              // proc4
};

extern "C" void kernel_launch(void* const* d_in, const int* in_sizes, int n_in,
                              void* d_out, int out_size, void* d_ws, size_t ws_size,
                              hipStream_t stream) {
  (void)in_sizes; (void)out_size; (void)ws_size;
  const float* x_low = (const float*)d_in[0];
  const float* x9    = (const float*)d_in[1];
  const float* x25   = (const float*)d_in[2];
  const float* xh    = (const float*)d_in[3];
  const float* z_std = (const float*)d_in[4];
  const int*   eL9   = (const int*)d_in[5];
  const int*   e925  = (const int*)d_in[6];
  const int*   e25h  = (const int*)d_in[7];
  const int*   ehh   = (const int*)d_in[8];

  const float* P[66];
  if (n_in >= 9 + 66) {
    for (int i = 0; i < 66; ++i) P[i] = (const float*)d_in[9 + i];
  } else {
    const float* base = (const float*)d_in[9];
    size_t off = 0;
    for (int i = 0; i < 66; ++i) { P[i] = base + off; off += kParamSizes[i]; }
  }
  const float *cb = P[0], *cbt = P[1], *cg = P[2], *cw = P[3];

  // ---- workspace carve ----
  char* wsb = (char*)d_ws;
  size_t off = 0;
  auto alloc = [&](size_t nbytes) -> void* {
    void* p = (void*)(wsb + off);
    off = (off + nbytes + 255) & ~(size_t)255;
    return p;
  };
  float* cA    = (float*)alloc(sizeof(float) * N_LOW * 125);
  float* cB    = (float*)alloc(sizeof(float) * N_LOW * 125);
  float* cst   = (float*)alloc(sizeof(float) * 16);
  float* enc45 = (float*)alloc(sizeof(float) * N_LOW * 45);
  float* h9    = (float*)alloc(sizeof(float) * N9 * 64);
  float* h25   = (float*)alloc(sizeof(float) * N25 * 64);
  float* hH    = (float*)alloc(sizeof(float) * NH * 64);
  float* xcat  = (float*)alloc(sizeof(float) * NH * 65);
  float* xpA   = (float*)alloc(sizeof(float) * NH * 128);
  float* xpB   = (float*)alloc(sizeof(float) * NH * 128);
  GatBufs t;
  t.xl   = (float*)alloc(sizeof(float) * NH * 128);       // max Ns*HC (7.68M f)
  t.xr   = (float*)alloc(sizeof(float) * NH * 128);       // covers 100000*64 too
  t.agg  = (float*)alloc(sizeof(float) * NH * 128);
  t.elog = (float*)alloc(sizeof(float) * 1400000);
  t.m    = (float*)alloc(sizeof(float) * 200000);
  t.ssum = (float*)alloc(sizeof(float) * 200000);
  t.cnt  = (float*)alloc(sizeof(float) * 120000);
  t.abf  = (bf16_t*)alloc(sizeof(bf16_t) * 8000000);      // max M*Kp (60000*128)
  t.abf2 = (bf16_t*)alloc(sizeof(bf16_t) * 3300000);      // max Nd*32
  t.wbf  = (bf16_t*)alloc(sizeof(bf16_t) * 16384);
  t.wbf2 = (bf16_t*)alloc(sizeof(bf16_t) * 16384);
  int* src2 = (int*)alloc(sizeof(int) * (EHH + NH));
  int* dst2 = (int*)alloc(sizeof(int) * (EHH + NH));
  float* bstats = (float*)alloc(sizeof(float) * 256);

  // ---- CNN encoder: 3x (depthwise conv + batch BN + ReLU), pool, flatten ----
  const float* cin = x_low;
  float* bufs[3] = { cA, cB, cA };
  for (int l = 0; l < 3; ++l) {
    float* y = bufs[l];
    k_fill<<<1, 16, 0, stream>>>(cst, 0.0f, 16);
    int tot = N_LOW * 125;
    k_dwconv<<<(tot + WG - 1) / WG, WG, 0, stream>>>(cin, cw + l * 45, cb + l * 5, y, cst, N_LOW);
    k_cnn_bn<<<(tot + WG - 1) / WG, WG, 0, stream>>>(y, cst, cg + l * 5, cbt + l * 5, N_LOW);
    cin = y;
  }
  k_pool45<<<(N_LOW * 45 + WG - 1) / WG, WG, 0, stream>>>(cA, enc45, N_LOW);

  // ---- downscale GATs: low->9x->25x->high ----
  run_gat(stream, enc45, N_LOW, 45, x9,  N9,  1, eL9,  eL9 + E1,  E1,
          P[4], P[8], P[5], P[9], P[6], P[7], 1, 0, h9, t);
  run_gat(stream, h9, N9, 64, x25, N25, 1, e925, e925 + E2, E2,
          P[10], P[14], P[11], P[15], P[12], P[13], 1, 0, h25, t);
  run_gat(stream, h25, N25, 64, xh, NH, 1, e25h, e25h + E3, E3,
          P[16], P[20], P[17], P[21], P[18], P[19], 1, 0, hH, t);

  // ---- concat z_std + self-loop edge list ----
  k_concat<<<(NH * 65 + WG - 1) / WG, WG, 0, stream>>>(z_std, hH, xcat, NH);
  int Etot = EHH + NH;
  k_make_edges<<<(Etot + WG - 1) / WG, WG, 0, stream>>>(ehh, src2, dst2, EHH, NH);

  // ---- processor GATs (2 heads x 64) + BN + ReLU ----
  const float* cur = xcat;
  int F = 65;
  for (int k = 0; k < 4; ++k) {
    float* gout = (k % 2 == 0) ? xpA : xpB;
    run_gat(stream, cur, NH, F, cur, NH, F, src2, dst2, Etot,
            P[36 + 6 * k], P[36 + 6 * k + 4], P[36 + 6 * k + 1], P[36 + 6 * k + 5],
            P[36 + 6 * k + 2], P[36 + 6 * k + 3], 2, 0, gout, t);
    k_fill<<<1, 256, 0, stream>>>(bstats, 0.0f, 256);
    k_colstats<<<256, 128, 0, stream>>>(gout, bstats, NH, 128);
    k_bn_apply_relu<<<(NH * 128 + WG - 1) / WG, WG, 0, stream>>>(
        gout, bstats, P[23 + 2 * k], P[22 + 2 * k], NH, 128);
    cur = gout;
    F = 128;
  }
  // final processor layer (1 head), ReLU in finish
  run_gat(stream, cur, NH, 128, cur, NH, 128, src2, dst2, Etot,
          P[60], P[64], P[61], P[65], P[62], P[63], 1, 1, xpA, t);

  // ---- predictor MLP ----
  gemm_f32(stream, xpA, NH, 64, P[30], 64, P[33], 1, xpB, t.abf, t.wbf);  // W1+b1, relu
  gemm_f32(stream, xpB, NH, 64, P[31], 32, P[34], 1, xpA, t.abf, t.wbf);  // W2+b2, relu
  k_dotN<<<(NH + WG - 1) / WG, WG, 0, stream>>>(xpA, P[32], P[35], (float*)d_out, NH, 32);
}
